// RNN_46712064311325
// MI455X (gfx1250) — compile-verified, matrix operand order checked by reading, not run on hardware
//
#include <hip/hip_runtime.h>
#include <math.h>

// ---------------------------------------------------------------------------
// RNN on MI455X (gfx1250), bf16 WMMA with f32 accumulation.
//
// Batch-parallel persistent tiles: each workgroup owns BM=32 batch rows,
// keeps h (bf16) in LDS across all 32 timesteps, and streams the L2-resident
// (8.4MB bf16) recurrent weight each step. No inter-WG sync needed.
// ---------------------------------------------------------------------------

typedef __attribute__((ext_vector_type(16))) __bf16 v16bf;
typedef __attribute__((ext_vector_type(8)))  __bf16 v8bf;
typedef __attribute__((ext_vector_type(8)))  float  v8f;

#define B_TOTAL 4096
#define XN 1024
#define HN 2048
#define YN 512
#define TSTEPS 32

#define BM 32                    // batch rows per workgroup (2 M-tiles)
#define NWAVES 16
#define NTHREADS (NWAVES * 32)
#define HPAD 8                   // bf16 elements of row padding (16B) vs LDS banks
#define HROWLEN (HN + HPAD)      // LDS row length in bf16 elements

// Cheap tanh: h is rounded to bf16 (~3e-3) immediately, so ~1e-6 accuracy of
// the exp/rcp formulation is far below the quantization already accepted.
// exp(2x)->inf  => rcp->0   => +1 ;  exp(2x)->0 => 1-2*1 = -1  (correct sat).
__device__ __forceinline__ float fast_tanh(float x) {
    float e = __expf(2.0f * x);                     // v_exp_f32
    float r = __builtin_amdgcn_rcpf(e + 1.0f);      // v_rcp_f32
    return 1.0f - 2.0f * r;                         // v_fma
}

// ---------------------------------------------------------------------------
// One dense layer computed by the whole workgroup:
//   Hs[BM, K] (bf16, LDS)  x  Wt[N, K] (bf16, global, pre-transposed)
//   bias folded into accumulator init; optional tanh; write back to LDS
//   (bf16) or to global out (f32).
// ---------------------------------------------------------------------------
template <int N_, int K_, bool TANH_, bool TO_LDS_>
__device__ __forceinline__ void layer(const __bf16* __restrict__ Wt,
                                      const float*  __restrict__ bias,
                                      __bf16* Hs,
                                      float* __restrict__ outg, // row stride YN
                                      int lane, int wave)
{
    constexpr int NPERWAVE = N_ / (NWAVES / 2);   // columns per wave
    constexpr int NT       = NPERWAVE / 16;       // 16x16 output tiles per wave

    const int mtile = wave & 1;
    const int nbase = (wave >> 1) * NPERWAVE;
    const int half  = lane >> 4;                  // 0: lanes 0-15, 1: lanes 16-31
    const int lrow  = lane & 15;

    // 16-bit fragment (16x32): lane half selects K sub-chunks {+0,+16} / {+8,+24}
    const int koff0 = 8 * half;
    const int koff1 = 16 + 8 * half;

    const __bf16* arow = Hs + (mtile * 16 + lrow) * HROWLEN;

    // C/D layout: N = lane%16 for every element of a tile -> bias is a splat.
    v8f acc[NT];
#pragma unroll
    for (int t = 0; t < NT; ++t) {
        const float bv = bias[nbase + t * 16 + lrow];
#pragma unroll
        for (int r = 0; r < 8; ++r) acc[t][r] = bv;
    }

    for (int k0 = 0; k0 < K_; k0 += 32) {
        v8bf a0 = *(const v8bf*)(arow + k0 + koff0);
        v8bf a1 = *(const v8bf*)(arow + k0 + koff1);
        v16bf a = __builtin_shufflevector(a0, a1, 0,1,2,3,4,5,6,7,8,9,10,11,12,13,14,15);
#pragma unroll
        for (int t = 0; t < NT; ++t) {
            const __bf16* brow = Wt + (size_t)(nbase + t * 16 + lrow) * K_;
            v8bf b0 = *(const v8bf*)(brow + k0 + koff0);
            v8bf b1 = *(const v8bf*)(brow + k0 + koff1);
            v16bf bb = __builtin_shufflevector(b0, b1, 0,1,2,3,4,5,6,7,8,9,10,11,12,13,14,15);
            acc[t] = __builtin_amdgcn_wmma_f32_16x16x32_bf16(
                false, a, false, bb, (short)0, acc[t], false, false);
        }
    }

    __syncthreads();   // everyone is done READING Hs before anyone overwrites it

#pragma unroll
    for (int t = 0; t < NT; ++t) {
        const int n = nbase + t * 16 + lrow;             // C/D: N = lane%16
#pragma unroll
        for (int r = 0; r < 8; ++r) {
            const int mrow = mtile * 16 + r + 8 * half;  // C/D: M = r + 8*(lane>=16)
            float v = acc[t][r];
            if (TANH_) v = fast_tanh(v);
            if (TO_LDS_) Hs[mrow * HROWLEN + n] = (__bf16)v;
            else         outg[(size_t)mrow * YN + n] = v;
        }
    }

    __syncthreads();   // new Hs visible before the next layer reads it
}

// ---------------------------------------------------------------------------
// One-time convert+transpose: out[n*K + k] = (bf16) in[k*N + n]
// 2D grid (no div/mod): x = column n (coalesced read), y = row k.
// ---------------------------------------------------------------------------
__global__ void transpose_bf16_kernel(const float* __restrict__ in,
                                      __bf16* __restrict__ out,
                                      int K, int N)
{
    int n = blockIdx.x * blockDim.x + threadIdx.x;
    int k = blockIdx.y;
    if (n < N) out[(size_t)n * K + k] = (__bf16)in[(size_t)k * N + n];
}

// ---------------------------------------------------------------------------
// Persistent-tile RNN kernel: grid.x = 4096/BM = 128 workgroups.
// ---------------------------------------------------------------------------
__global__ __launch_bounds__(NTHREADS)
void rnn_persistent_kernel(const float* __restrict__ x,
                           const float* __restrict__ b,
                           const float* __restrict__ b_out,
                           const __bf16* __restrict__ w_in_t,   // [HN][XN]
                           const __bf16* __restrict__ w_t,      // [HN][HN]
                           const __bf16* __restrict__ w_out_t,  // [YN][HN]
                           float* __restrict__ out)
{
    __shared__ __align__(16) __bf16 Hs[BM * HROWLEN];   // ~128.5 KB

    const int m0   = blockIdx.x * BM;
    const int tid  = (int)threadIdx.x;
    const int lane = tid & 31;
    const int wave = tid >> 5;

    // Stage this WG's x tile into LDS as bf16 (cols 0..XN-1 of Hs).
    for (int i = tid; i < BM * XN; i += NTHREADS) {
        int r = i / XN, c = i % XN;
        Hs[r * HROWLEN + c] = (__bf16)x[(size_t)(m0 + r) * XN + c];
    }
    __syncthreads();

    // Input projection: h = tanh(x @ W_in + b)
    layer<HN, XN, true, true>(w_in_t, b, Hs, nullptr, lane, wave);

    // 31 recurrent steps: h = tanh(h @ W + b)   (W stays hot in L2)
    for (int t = 0; t < TSTEPS - 1; ++t)
        layer<HN, HN, true, true>(w_t, b, Hs, nullptr, lane, wave);

    // Readout: out = h @ W_out + b_out   (f32 to global)
    layer<YN, HN, false, false>(w_out_t, b_out, Hs,
                                out + (size_t)m0 * YN, lane, wave);
}

// ---------------------------------------------------------------------------
// Workspace layout (needs 14 MB):
//   [0,            4MB)  w_in_t : 2048x1024 bf16
//   [4MB,         12MB)  w_t    : 2048x2048 bf16
//   [12MB,        14MB)  w_out_t:  512x2048 bf16
// ---------------------------------------------------------------------------
extern "C" void kernel_launch(void* const* d_in, const int* in_sizes, int n_in,
                              void* d_out, int out_size, void* d_ws, size_t ws_size,
                              hipStream_t stream)
{
    const float* x     = (const float*)d_in[0];
    const float* w_in  = (const float*)d_in[1];
    const float* w     = (const float*)d_in[2];
    const float* b     = (const float*)d_in[3];
    const float* w_out = (const float*)d_in[4];
    const float* b_out = (const float*)d_in[5];
    float* out = (float*)d_out;

    char* ws = (char*)d_ws;
    __bf16* w_in_t  = (__bf16*)(ws);
    __bf16* w_t     = (__bf16*)(ws + (size_t)XN * HN * 2);
    __bf16* w_out_t = (__bf16*)(ws + (size_t)XN * HN * 2 + (size_t)HN * HN * 2);

    const int tpb = 256;
    dim3 g1((HN + tpb - 1) / tpb, XN);
    transpose_bf16_kernel<<<g1, tpb, 0, stream>>>(w_in,  w_in_t,  XN, HN);
    dim3 g2((HN + tpb - 1) / tpb, HN);
    transpose_bf16_kernel<<<g2, tpb, 0, stream>>>(w,     w_t,     HN, HN);
    dim3 g3((YN + tpb - 1) / tpb, HN);
    transpose_bf16_kernel<<<g3, tpb, 0, stream>>>(w_out, w_out_t, HN, YN);

    rnn_persistent_kernel<<<B_TOTAL / BM, NTHREADS, 0, stream>>>(
        x, b, b_out, w_in_t, w_t, w_out_t, out);
}
